// ReasoningGAT_53970559041909
// MI455X (gfx1250) — compile-verified, matrix operand order checked by reading, not run on hardware
//
#include <hip/hip_runtime.h>
#include <hip/hip_bf16.h>

#define TB 256

typedef __attribute__((ext_vector_type(16))) __bf16 v16bf;
typedef __attribute__((ext_vector_type(8)))  float  v8f;
typedef __attribute__((ext_vector_type(4)))  unsigned int u32x4;
typedef __attribute__((ext_vector_type(2)))  float  f32x2;
typedef __attribute__((ext_vector_type(4)))  float  f32x4;

// ---------------------------------------------------------------------------
// helpers
// ---------------------------------------------------------------------------
// single v_cvt_pk_bf16_f32 (RNE) -- compiler doesn't pattern-match this pack
__device__ __forceinline__ unsigned int pack2_bf16(float x, float y) {
    unsigned int r;
    asm("v_cvt_pk_bf16_f32 %0, %1, %2" : "=v"(r) : "v"(x), "v"(y));
    return r;
}

__device__ __forceinline__ v16bf load_frag(const unsigned int* row, int pbase) {
    union { u32x4 q[2]; v16bf v; } f;
    f.q[0] = *(const u32x4*)(row + pbase);      // ds_load_b128
    f.q[1] = *(const u32x4*)(row + pbase + 8);  // ds_load_b128
    return f.v;
}

__device__ __forceinline__ void atomicAddF(float* addr, float v) {
    __hip_atomic_fetch_add(addr, v, __ATOMIC_RELAXED, __HIP_MEMORY_SCOPE_AGENT);
}

// float segment-max without CAS loop: int-max for >=0, uint-min for <0
__device__ __forceinline__ void atomicMaxF(float* addr, float v) {
    if (v >= 0.f)
        __hip_atomic_fetch_max((int*)addr, __float_as_int(v),
                               __ATOMIC_RELAXED, __HIP_MEMORY_SCOPE_AGENT);
    else
        __hip_atomic_fetch_min((unsigned int*)addr,
                               (unsigned int)__float_as_int(v),
                               __ATOMIC_RELAXED, __HIP_MEMORY_SCOPE_AGENT);
}

__device__ __forceinline__ void edge_sd(const long long* __restrict__ ei,
                                        int e, int E, int& s, int& d) {
    if (e < E) { s = (int)ei[e]; d = (int)ei[E + e]; }
    else       { s = d = e - E; }                       // self loops appended
}

__device__ __forceinline__ int imin(int a, int b) { return a < b ? a : b; }

// ---------------------------------------------------------------------------
// bf16-WMMA GEMM: C[M,N] = A[M,K] x B[K,N], all fp32 in memory.
// Block = 8 waves, tile 128(M) x 64(N), K-step 64 (2 WMMA half-steps).
// Double-buffered LDS (packed bf16 pairs), two barriers per K-step.
// Wave (wm 0..3, wn 0..1) owns a 32x32 tile = 2x2 WMMA subtiles.
// ---------------------------------------------------------------------------
__global__ __launch_bounds__(TB)
void k_gemm_bf16(const float* __restrict__ A, const float* __restrict__ B,
                 float* __restrict__ C, int M, int K, int Nn) {
    __shared__ unsigned int As[2][128][32];  // [buf][m][K-pair]  2 x 16 KB
    __shared__ unsigned int Bs[2][64][32];   // [buf][n][K-pair]  2 x  8 KB

    const int tid   = threadIdx.x;
    const int lane  = tid & 31;
    const int wave  = tid >> 5;
    const int wm    = wave & 3;
    const int wn    = wave >> 2;
    const int fr    = lane & 15;          // fragment row (M) / col (N)
    const int pbase = (lane >> 4) << 2;   // K-pair base within half: 0 or 4
    const int blockM = blockIdx.x * 128;
    const int blockN = blockIdx.y * 64;

    // staging thread mapping
    const int pa  = tid & 31;             // A: pair column (0..31)
    const int ma0 = tid >> 5;             // A: row base, 8 rows per pass
    const int nb  = tid & 63;             // B: n
    const int kp0 = tid >> 6;             // B: pair base (0..3)

    v8f acc[2][2] = {};
    f32x2 aR[16];
    float b0R[8], b1R[8];

    // branch-free global staging: clamp row (bad rows only affect C rows >= M,
    // which the store guards), so loads batch without exec-mask juggling.
    auto loadG = [&](int k0) {
        #pragma unroll
        for (int it = 0; it < 16; ++it) {
            int gm = imin(blockM + ma0 + it * 8, M - 1);
            aR[it] = *(const f32x2*)(A + (size_t)gm * K + k0 + 2 * pa);
        }
        #pragma unroll
        for (int it = 0; it < 8; ++it) {
            int k = k0 + 2 * (kp0 + it * 4);
            b0R[it] = B[(size_t)k * Nn + blockN + nb];
            b1R[it] = B[(size_t)(k + 1) * Nn + blockN + nb];
        }
    };
    auto storeLDS = [&](int buf) {
        #pragma unroll
        for (int it = 0; it < 16; ++it)
            As[buf][ma0 + it * 8][pa] = pack2_bf16(aR[it].x, aR[it].y);
        #pragma unroll
        for (int it = 0; it < 8; ++it)
            Bs[buf][nb][kp0 + it * 4] = pack2_bf16(b0R[it], b1R[it]);
    };

    const int nk = K >> 6;
    loadG(0);
    for (int i = 0; i < nk; ++i) {
        const int buf = i & 1;
        storeLDS(buf);
        __syncthreads();
        if (i + 1 < nk) loadG((i + 1) << 6);
        #pragma unroll
        for (int kh = 0; kh < 2; ++kh) {
            const int pb = kh * 16 + pbase;
            v16bf fA0 = load_frag(As[buf][wm * 32 +      fr], pb);
            v16bf fA1 = load_frag(As[buf][wm * 32 + 16 + fr], pb);
            v16bf fB0 = load_frag(Bs[buf][wn * 32 +      fr], pb);
            v16bf fB1 = load_frag(Bs[buf][wn * 32 + 16 + fr], pb);
            acc[0][0] = __builtin_amdgcn_wmma_f32_16x16x32_bf16(
                false, fA0, false, fB0, (short)0, acc[0][0], false, false);
            acc[0][1] = __builtin_amdgcn_wmma_f32_16x16x32_bf16(
                false, fA0, false, fB1, (short)0, acc[0][1], false, false);
            acc[1][0] = __builtin_amdgcn_wmma_f32_16x16x32_bf16(
                false, fA1, false, fB0, (short)0, acc[1][0], false, false);
            acc[1][1] = __builtin_amdgcn_wmma_f32_16x16x32_bf16(
                false, fA1, false, fB1, (short)0, acc[1][1], false, false);
        }
        __syncthreads();
    }

    // ---- store D (f32 16x16 layout: lane = N + 16*(M>=8), vgpr r = M&7)
    #pragma unroll
    for (int mi = 0; mi < 2; ++mi) {
        #pragma unroll
        for (int ni = 0; ni < 2; ++ni) {
            int ncol  = blockN + wn * 32 + ni * 16 + fr;
            int mbase = blockM + wm * 32 + mi * 16 + ((lane >> 4) << 3);
            #pragma unroll
            for (int r = 0; r < 8; ++r) {
                int m = mbase + r;
                if (m < M) C[(size_t)m * Nn + ncol] = acc[mi][ni][r];
            }
        }
    }
}

// ---------------------------------------------------------------------------
// per-(node,head) attention scores: s = sum_c h[n,h,c]*a[h,c]  (warp reduce)
// ---------------------------------------------------------------------------
__global__ __launch_bounds__(TB)
void k_node_scores(const float* __restrict__ h, const float* __restrict__ asrc,
                   const float* __restrict__ adst, float* __restrict__ ss,
                   float* __restrict__ sd, int n, int H, int C) {
    int gw   = blockIdx.x * (TB / 32) + (threadIdx.x >> 5);
    int lane = threadIdx.x & 31;
    if (gw >= n * H) return;
    int node = gw / H, head = gw % H;
    const float* hp = h + (size_t)node * H * C + (size_t)head * C;
    const float* ap = asrc + (size_t)head * C;
    const float* bp = adst + (size_t)head * C;
    float v0 = 0.f, v1 = 0.f;
    for (int c = lane; c < C; c += 32) {
        float hv = hp[c];
        v0 += hv * ap[c];
        v1 += hv * bp[c];
    }
    #pragma unroll
    for (int o = 16; o > 0; o >>= 1) {
        v0 += __shfl_down(v0, o);
        v1 += __shfl_down(v1, o);
    }
    if (lane == 0) { ss[gw] = v0; sd[gw] = v1; }
}

__global__ void k_init_md(unsigned int* __restrict__ mbits,
                          float* __restrict__ dn, int n) {
    int i = blockIdx.x * blockDim.x + threadIdx.x;
    if (i < n) { mbits[i] = 0xFF800000u; dn[i] = 0.f; }   // m = -inf, denom = 0
}

__global__ void k_edge_logits_max(const long long* __restrict__ ei,
                                  const float* __restrict__ ss,
                                  const float* __restrict__ sd,
                                  float* __restrict__ ev, float* __restrict__ mx,
                                  int E, int ET, int H) {
    int t = blockIdx.x * blockDim.x + threadIdx.x;
    if (t >= ET * H) return;
    int e = t / H, h = t % H, s, d;
    edge_sd(ei, e, E, s, d);
    float v = ss[s * H + h] + sd[d * H + h];
    v = v > 0.f ? v : 0.2f * v;                // leaky_relu(0.2)
    ev[t] = v;
    atomicMaxF(&mx[d * H + h], v);
}

__global__ void k_m_guard(float* __restrict__ mx, int n) {
    int i = blockIdx.x * blockDim.x + threadIdx.x;
    if (i < n) {
        float v = mx[i];
        if ((__float_as_uint(v) & 0x7F800000u) == 0x7F800000u) mx[i] = 0.f;
    }
}

__global__ void k_edge_exp_sum(const long long* __restrict__ ei,
                               float* __restrict__ ev, const float* __restrict__ mx,
                               float* __restrict__ dn, int E, int ET, int H) {
    int t = blockIdx.x * blockDim.x + threadIdx.x;
    if (t >= ET * H) return;
    int e = t / H, h = t % H, s, d;
    edge_sd(ei, e, E, s, d);
    float p = expf(ev[t] - mx[d * H + h]);
    ev[t] = p;
    atomicAddF(&dn[d * H + h], p);
}

__global__ void k_zero(float* __restrict__ p, size_t n) {
    size_t i = (size_t)blockIdx.x * blockDim.x + threadIdx.x;
    size_t stride = (size_t)gridDim.x * blockDim.x;
    for (; i < n; i += stride) p[i] = 0.f;
}

// one block per edge: out[dst,h,c] += h[src,h,c] * alpha[e,h]
__global__ void k_edge_aggregate(const long long* __restrict__ ei,
                                 const float* __restrict__ hf,
                                 const float* __restrict__ pv,
                                 const float* __restrict__ dn,
                                 float* __restrict__ out,
                                 int E, int H, int cshift) {
    __shared__ float alpha_s[8];
    int e = blockIdx.x, s, d;
    edge_sd(ei, e, E, s, d);
    int tid = threadIdx.x;
    if (tid < H)
        alpha_s[tid] = pv[e * H + tid] / (dn[d * H + tid] + 1e-16f);
    __syncthreads();
    int HC = H << cshift;
    const float* src = hf + (size_t)s * HC;
    float* dst = out + (size_t)d * HC;
    for (int j = tid; j < HC; j += blockDim.x)
        atomicAddF(&dst[j], src[j] * alpha_s[j >> cshift]);
}

// out = act(in + bias[col]); mode 0=none 1=elu 2=relu
__global__ void k_bias_act(const float* __restrict__ in, const float* __restrict__ bias,
                           float* __restrict__ out, size_t total, int cols, int mode) {
    size_t i = (size_t)blockIdx.x * blockDim.x + threadIdx.x;
    size_t stride = (size_t)gridDim.x * blockDim.x;
    for (; i < total; i += stride) {
        float v = in[i] + bias[i % cols];
        if (mode == 1)      v = v > 0.f ? v : expm1f(v);
        else if (mode == 2) v = v > 0.f ? v : 0.f;
        out[i] = v;
    }
}

// scores[n] = dot(h[n,0:128], w) + b ; warp per node
__global__ __launch_bounds__(TB)
void k_final_dot(const float* __restrict__ h, const float* __restrict__ w,
                 const float* __restrict__ b, float* __restrict__ out, int n) {
    int node = blockIdx.x * (TB / 32) + (threadIdx.x >> 5);
    int lane = threadIdx.x & 31;
    if (node >= n) return;
    f32x4 hv = *(const f32x4*)(h + (size_t)node * 128 + lane * 4);
    f32x4 wv = *(const f32x4*)(w + lane * 4);
    float v = hv.x * wv.x + hv.y * wv.y + hv.z * wv.z + hv.w * wv.w;
    #pragma unroll
    for (int o = 16; o > 0; o >>= 1) v += __shfl_down(v, o);
    if (lane == 0) out[node] = v + b[0];
}

// ---------------------------------------------------------------------------
// host-side orchestration
// ---------------------------------------------------------------------------
static void launch_gemm(const float* A, const float* B, float* C,
                        int M, int K, int Nn, hipStream_t s) {
    dim3 g((M + 127) / 128, Nn / 64);
    k_gemm_bf16<<<g, TB, 0, s>>>(A, B, C, M, K, Nn);
}

static void launch_bias_act(const float* in, const float* bias, float* out,
                            int rows, int cols, int mode, hipStream_t s) {
    size_t total = (size_t)rows * cols;
    int blocks = (int)((total + TB - 1) / TB);
    if (blocks > 8192) blocks = 8192;
    k_bias_act<<<blocks, TB, 0, s>>>(in, bias, out, total, cols, mode);
}

static void gat_layer(const float* xin, int K, const float* W,
                      const float* a_s, const float* a_d,
                      int H, int C, int cshift,
                      const long long* ei, int E, int n,
                      float* hbuf, float* outbuf,
                      float* ss, float* sd, float* mx, float* dn, float* ev,
                      hipStream_t s) {
    const int HC = H * C, ET = E + n, tE = ET * H;
    launch_gemm(xin, W, hbuf, n, K, HC, s);
    k_node_scores<<<(n * H + 7) / 8, TB, 0, s>>>(hbuf, a_s, a_d, ss, sd, n, H, C);
    k_init_md<<<(n * H + TB - 1) / TB, TB, 0, s>>>((unsigned int*)mx, dn, n * H);
    k_edge_logits_max<<<(tE + TB - 1) / TB, TB, 0, s>>>(ei, ss, sd, ev, mx, E, ET, H);
    k_m_guard<<<(n * H + TB - 1) / TB, TB, 0, s>>>(mx, n * H);
    k_edge_exp_sum<<<(tE + TB - 1) / TB, TB, 0, s>>>(ei, ev, mx, dn, E, ET, H);
    {
        size_t total = (size_t)n * HC;
        int blocks = (int)((total + TB - 1) / TB);
        if (blocks > 8192) blocks = 8192;
        k_zero<<<blocks, TB, 0, s>>>(outbuf, total);
    }
    int aggT = HC < TB ? HC : TB;   // conv3: 64 threads, others 256
    k_edge_aggregate<<<ET, aggT, 0, s>>>(ei, hbuf, ev, dn, outbuf, E, H, cshift);
}

extern "C" void kernel_launch(void* const* d_in, const int* in_sizes, int n_in,
                              void* d_out, int out_size, void* d_ws, size_t ws_size,
                              hipStream_t stream) {
    (void)n_in; (void)out_size; (void)ws_size;
    const float*     x   = (const float*)d_in[0];
    const long long* ei  = (const long long*)d_in[1];
    const float* W1  = (const float*)d_in[2];
    const float* as1 = (const float*)d_in[3];
    const float* ad1 = (const float*)d_in[4];
    const float* b1  = (const float*)d_in[5];
    const float* W2  = (const float*)d_in[6];
    const float* as2 = (const float*)d_in[7];
    const float* ad2 = (const float*)d_in[8];
    const float* b2  = (const float*)d_in[9];
    const float* W3  = (const float*)d_in[10];
    const float* as3 = (const float*)d_in[11];
    const float* ad3 = (const float*)d_in[12];
    const float* b3  = (const float*)d_in[13];
    const float* r1w = (const float*)d_in[14];
    const float* r1b = (const float*)d_in[15];
    const float* r2w = (const float*)d_in[16];
    const float* r2b = (const float*)d_in[17];
    const float* r3w = (const float*)d_in[18];
    const float* r3b = (const float*)d_in[19];

    const int n = in_sizes[0] / 768;   // 10000
    const int E = in_sizes[1] / 2;     // 80000

    // carve workspace
    size_t off = 0;
    auto alloc = [&](size_t bytes) -> float* {
        float* p = (float*)((char*)d_ws + off);
        off += (bytes + 255) & ~(size_t)255;
        return p;
    };
    const size_t big = (size_t)n * 2048 * sizeof(float);
    float* RA = alloc(big);                          // h buffers
    float* RB = alloc(big);                          // aggregate buffers
    float* RC = alloc(big);                          // activations
    float* ss = alloc((size_t)n * 8 * sizeof(float));
    float* sd = alloc((size_t)n * 8 * sizeof(float));
    float* mx = alloc((size_t)n * 8 * sizeof(float));
    float* dn = alloc((size_t)n * 8 * sizeof(float));
    float* ev = alloc((size_t)(E + n) * 8 * sizeof(float));

    float* scores = (float*)d_out;     // [n]
    float* x3     = scores + n;        // [n,64]

    // conv1: 768 -> 8 x 256, ELU
    gat_layer(x,  768,  W1, as1, ad1, 8, 256, 8, ei, E, n, RA, RB, ss, sd, mx, dn, ev, stream);
    launch_bias_act(RB, b1, RC, n, 2048, 1, stream);
    // conv2: 2048 -> 4 x 256, ELU
    gat_layer(RC, 2048, W2, as2, ad2, 4, 256, 8, ei, E, n, RA, RB, ss, sd, mx, dn, ev, stream);
    launch_bias_act(RB, b2, RC, n, 1024, 1, stream);
    // conv3: 1024 -> 1 x 64 (aggregate straight into d_out's x3 region)
    gat_layer(RC, 1024, W3, as3, ad3, 1, 64, 6, ei, E, n, RA, x3, ss, sd, mx, dn, ev, stream);
    launch_bias_act(x3, b3, x3, n, 64, 0, stream);
    // reasoning MLP: 64 -> 256 (relu) -> 128 (relu) -> 1
    launch_gemm(x3, r1w, RA, n, 64, 256, stream);
    launch_bias_act(RA, r1b, RB, n, 256, 2, stream);
    launch_gemm(RB, r2w, RA, n, 256, 128, stream);
    launch_bias_act(RA, r2b, RC, n, 128, 2, stream);
    k_final_dot<<<(n + 7) / 8, TB, 0, stream>>>(RC, r3w, r3b, scores, n);
}
